// SparseLinear_16965120819979
// MI455X (gfx1250) — matrix operand order repaired and checked by reading, not compiled, not run on hardware
//
#include <hip/hip_runtime.h>
#include <hip/hip_bf16.h>

// Problem constants (match reference)
#define IN_DIM  4096
#define OUT_DIM 4096
#define M_DIM   (8 * 2048)      // B*S
#define K_DIM   IN_DIM
#define N_DIM   OUT_DIM
#define TOTAL   (OUT_DIM * IN_DIM)

// GEMM tiling
#define BM 128
#define BN 128
#define BK 32

typedef __bf16          v16bf __attribute__((ext_vector_type(16)));
typedef float           v8f   __attribute__((ext_vector_type(8)));
typedef unsigned short  v8us  __attribute__((ext_vector_type(8)));
typedef unsigned int    u32x4 __attribute__((ext_vector_type(4)));
typedef unsigned int    u32x8 __attribute__((ext_vector_type(8)));

// ---------------- workspace layout (u32 units for the selection state) ----
// [0..2047]        histW1   (bits 30:20)
// [2048..6143]     histW2   (bits 19:8)
// [6144..6399]     histW3   (bits 7:0)
// [6400..8447]     histG1
// [8448..12543]    histG2
// [12544..12799]   histG3
// [12800..12815]   res:  0:d1 1:rem 2:d2 3:rem 4:d3 5:rem  6:g1 7:rem 8:g2 9:rem 10:g3 11:rem
// byte 65536:                whi plane (TOTAL ushort, 32 MB)
// byte 65536 + 2*TOTAL:      wlo plane (TOTAL ushort, 32 MB)
#define OFF_HW1  0
#define OFF_HW2  2048
#define OFF_HW3  6144
#define OFF_HG1  6400
#define OFF_HG2  8448
#define OFF_HG3  12544
#define OFF_RES  12800
#define OFF_PLANES 65536u

__device__ __forceinline__ unsigned absbits(float f) {
  return __float_as_uint(f) & 0x7FFFFFFFu;
}
__device__ __forceinline__ unsigned short bf16_rne(float f) {
  unsigned u = __float_as_uint(f);
  unsigned r = u + 0x7FFFu + ((u >> 16) & 1u);
  return (unsigned short)(r >> 16);
}
// Generic LDS pointers keep the LDS byte offset in the low 32 bits
// (ISA 10.2).
__device__ __forceinline__ unsigned lds_offset(const void* p) {
  return (unsigned)(size_t)p;
}

// --------------------------------------------------------------------------
// Tensor Data Mover: DMA one 2D tile of a row-major bf16 tensor into LDS.
// D# group0: count=1 | lds_addr | global_addr[56:0] | type=2
// D# group1: data_size=1 (2B), tensor dims/strides, tile dims (2D: VADDR2/3
//            omitted -> NULL). Tracked with TENSORcnt.
// --------------------------------------------------------------------------
__device__ __forceinline__ void tdm_load_2d_bf16(const unsigned short* gsrc,
                                                 unsigned lds_addr,
                                                 unsigned tensor_d0,
                                                 unsigned tensor_d1,
                                                 unsigned tile_d0,
                                                 unsigned tile_d1,
                                                 unsigned stride0) {
  unsigned long long ga = (unsigned long long)gsrc;
  u32x4 g0;
  g0.x = 1u;                                            // count=1, user mode
  g0.y = lds_addr;                                      // bits 63:32
  g0.z = (unsigned)(ga & 0xFFFFFFFFu);                  // global_addr[31:0]
  g0.w = (unsigned)((ga >> 32) & 0x01FFFFFFu) | (2u << 30);  // addr[56:32]|type=2
  u32x8 g1;
  g1.s0 = 1u << 16;                                     // data_size=1 -> 2 bytes
  g1.s1 = (tensor_d0 & 0xFFFFu) << 16;                  // tensor_dim0[15:0]
  g1.s2 = ((tensor_d0 >> 16) & 0xFFFFu) | ((tensor_d1 & 0xFFFFu) << 16);
  g1.s3 = ((tensor_d1 >> 16) & 0xFFFFu) | ((tile_d0 & 0xFFFFu) << 16);
  g1.s4 = tile_d1 & 0xFFFFu;                            // tile_dim1; tile_dim2=0
  g1.s5 = stride0;                                      // tensor_dim0_stride[31:0]
  g1.s6 = 0u;                                           // stride0[47:32], stride1 lo
  g1.s7 = 0u;                                           // stride1 hi
  asm volatile("tensor_load_to_lds %0, %1" ::"s"(g0), "s"(g1) : "memory");
}

// --------------------------------------------------------------------------
// Multi-level radix-select histogram. level 1: bits30:20 (2048 bins),
// level 2: bits19:8 (4096 bins, restricted to winning level-1 bucket),
// level 3: bits7:0 (256 bins, restricted to winning 23-bit prefix).
// If `mask` != null this is a grad pass: only count "grow candidates",
// i.e. slots that are masked-out after the (already resolved) drop step.
// --------------------------------------------------------------------------
__global__ void hist_level_kernel(const float* __restrict__ vals,
                                  unsigned* __restrict__ hist,
                                  const unsigned* __restrict__ res,
                                  int nbins, int level, int resBase,
                                  const float* __restrict__ wcand,
                                  const unsigned char* __restrict__ mask) {
  __shared__ unsigned sh[4096];
  for (int b = threadIdx.x; b < nbins; b += blockDim.x) sh[b] = 0u;
  __syncthreads();

  unsigned b1 = 0u, match23 = 0u, Kd = 0u;
  if (level >= 2) b1 = res[resBase];
  if (level == 3) match23 = (res[resBase] << 12) | res[resBase + 2];
  if (mask) Kd = (res[0] << 20) | (res[2] << 8) | res[4];

  const int stride = gridDim.x * blockDim.x;
  for (int i = blockIdx.x * blockDim.x + threadIdx.x; i < TOTAL; i += stride) {
    if (mask) {  // grow candidate = !mask || dropped
      bool cand = (!mask[i]) || (absbits(wcand[i]) < Kd);
      if (!cand) continue;
    }
    unsigned key = absbits(vals[i]);
    if (level == 1) {
      atomicAdd(&sh[key >> 20], 1u);
    } else if (level == 2) {
      if ((key >> 20) == b1) atomicAdd(&sh[(key >> 8) & 0xFFFu], 1u);
    } else {
      if ((key >> 8) == match23) atomicAdd(&sh[key & 0xFFu], 1u);
    }
  }
  __syncthreads();
  for (int b = threadIdx.x; b < nbins; b += blockDim.x)
    if (sh[b]) atomicAdd(&hist[b], sh[b]);
}

// Find the bucket containing the k-th element (ascending or descending scan).
__global__ void select_kernel(unsigned* __restrict__ res,
                              const unsigned* __restrict__ hist,
                              int nbins, int desc,
                              const int* __restrict__ ndrop, int kResIdx,
                              int outB, int outR) {
  if (threadIdx.x != 0 || blockIdx.x != 0) return;
  long k = (kResIdx >= 0) ? (long)res[kResIdx] : (long)ndrop[0];
  if (k <= 0) { res[outB] = desc ? 0xFFFFFFFFu : 0u; res[outR] = 0u; return; }
  long cum = 0;
  if (!desc) {
    for (int b = 0; b < nbins; ++b) {
      long c = cum + (long)hist[b];
      if (c >= k) { res[outB] = (unsigned)b; res[outR] = (unsigned)(k - cum); return; }
      cum = c;
    }
    res[outB] = (unsigned)(nbins - 1); res[outR] = 0u;
  } else {
    for (int b = nbins - 1; b >= 0; --b) {
      long c = cum + (long)hist[b];
      if (c >= k) { res[outB] = (unsigned)b; res[outR] = (unsigned)(k - cum); return; }
      cum = c;
    }
    res[outB] = 0u; res[outR] = 0u;
  }
}

// Apply drop/grow thresholds, materialize masked weight as bf16 hi/lo planes.
__global__ void finalize_kernel(const float* __restrict__ w,
                                const float* __restrict__ g,
                                const unsigned char* __restrict__ mask,
                                const unsigned* __restrict__ res,
                                unsigned short* __restrict__ whi,
                                unsigned short* __restrict__ wlo) {
  const unsigned Kd = (res[0] << 20) | (res[2] << 8) | res[4];
  const unsigned Kg = (res[6] << 20) | (res[8] << 8) | res[10];
  const int stride = gridDim.x * blockDim.x;
  for (int i = blockIdx.x * blockDim.x + threadIdx.x; i < TOTAL; i += stride) {
    float wv = w[i];
    bool dropped = absbits(wv) < Kd;
    bool flat = mask[i] && !dropped;
    bool grown = (!flat) && (absbits(g[i]) > Kg);
    float mw = (flat || grown) ? wv : 0.0f;
    unsigned short h = bf16_rne(mw);
    float hf = __uint_as_float((unsigned)h << 16);
    whi[i] = h;
    wlo[i] = bf16_rne(mw - hf);
  }
}

// --------------------------------------------------------------------------
// Compute-bound GEMM: Y[M,N] = X[M,K] * W[N,K]^T using bf16 hi/lo split
// (3x v_wmma_f32_16x16x32_bf16 per 16x16x32 fp32-accurate MAC block).
// W tiles arrive via the Tensor Data Mover; X is converted fp32->bf16 hi/lo
// in-flight.
// --------------------------------------------------------------------------
__global__ __launch_bounds__(256, 1)
void wmma_gemm_kernel(const float* __restrict__ X,
                      const unsigned short* __restrict__ Whi,
                      const unsigned short* __restrict__ Wlo,
                      float* __restrict__ Y) {
  __shared__ unsigned short sXh[BM][BK];
  __shared__ unsigned short sXl[BM][BK];
  __shared__ unsigned short sWh[BN][BK];
  __shared__ unsigned short sWl[BN][BK];

  const int tid  = threadIdx.x;
  const int wid  = tid >> 5;
  const int lane = tid & 31;
  const int half = lane >> 4;   // ISA half-wave: selects K-phase of fragments
  const int lr   = lane & 15;
  const int wave_m = wid & 3;   // 4 waves along M, 32 rows each
  const int wave_n = wid >> 2;  // 2 waves along N, 64 cols each
  const int m0 = blockIdx.y * BM;
  const int n0 = blockIdx.x * BN;

  const unsigned sWh_off = lds_offset(&sWh[0][0]);
  const unsigned sWl_off = lds_offset(&sWl[0][0]);

  v8f acc[2][4];
  const v8f zero = {0.f, 0.f, 0.f, 0.f, 0.f, 0.f, 0.f, 0.f};
#pragma unroll
  for (int a = 0; a < 2; ++a)
#pragma unroll
    for (int b = 0; b < 4; ++b) acc[a][b] = zero;

  for (int k0 = 0; k0 < K_DIM; k0 += BK) {
    // ---- W hi/lo tiles: one TDM descriptor each (128x32 bf16 2D tile).
    // TDM ignores EXEC, so gate on a *scalar* branch: only wave 0 issues.
    if (__builtin_amdgcn_readfirstlane(wid) == 0) {
      tdm_load_2d_bf16(Whi + (size_t)n0 * K_DIM + k0, sWh_off,
                       K_DIM, OUT_DIM, BK, BN, K_DIM);
      tdm_load_2d_bf16(Wlo + (size_t)n0 * K_DIM + k0, sWl_off,
                       K_DIM, OUT_DIM, BK, BN, K_DIM);
    }

    // ---- stage X tile: fp32 -> bf16 hi/lo in LDS (1024 float4 over 256 thr)
#pragma unroll
    for (int j = 0; j < 4; ++j) {
      int q = tid + 256 * j;
      int r = q >> 3;
      int c = (q & 7) << 2;
      const float4 v = *(const float4*)(X + (size_t)(m0 + r) * K_DIM + k0 + c);
      float vv[4] = {v.x, v.y, v.z, v.w};
#pragma unroll
      for (int e = 0; e < 4; ++e) {
        unsigned short h = bf16_rne(vv[e]);
        sXh[r][c + e] = h;
        sXl[r][c + e] = bf16_rne(vv[e] - __uint_as_float((unsigned)h << 16));
      }
    }
    if (k0 + BK < K_DIM) {  // global_prefetch_b8 of next X K-tile
      __builtin_prefetch(X + (size_t)(m0 + (tid >> 1)) * K_DIM + k0 + BK, 0, 1);
    }
    __builtin_amdgcn_s_wait_tensorcnt(0);  // wave0: W tiles landed in LDS
    __syncthreads();

    // ---- A fragments (ISA 16-bit A 16x32 layout:
    //      lanes 0-15: K 0-7 & 16-23; lanes 16-31: K 8-15 & 24-31)
    v16bf axh[2], axl[2];
#pragma unroll
    for (int mt = 0; mt < 2; ++mt) {
      int row = wave_m * 32 + mt * 16 + lr;
      v8us a0 = *(const v8us*)&sXh[row][half * 8];
      v8us a1 = *(const v8us*)&sXh[row][16 + half * 8];
      axh[mt] = __builtin_bit_cast(
          v16bf, __builtin_shufflevector(a0, a1, 0, 1, 2, 3, 4, 5, 6, 7, 8, 9,
                                         10, 11, 12, 13, 14, 15));
      v8us b0 = *(const v8us*)&sXl[row][half * 8];
      v8us b1 = *(const v8us*)&sXl[row][16 + half * 8];
      axl[mt] = __builtin_bit_cast(
          v16bf, __builtin_shufflevector(b0, b1, 0, 1, 2, 3, 4, 5, 6, 7, 8, 9,
                                         10, 11, 12, 13, 14, 15));
    }

    // ---- B fragments (lane = one N column, same K split as A) + WMMA chain
#pragma unroll
    for (int nt = 0; nt < 4; ++nt) {
      int coln = wave_n * 64 + nt * 16 + lr;
      v8us c0 = *(const v8us*)&sWh[coln][half * 8];
      v8us c1 = *(const v8us*)&sWh[coln][16 + half * 8];
      v16bf bh = __builtin_bit_cast(
          v16bf, __builtin_shufflevector(c0, c1, 0, 1, 2, 3, 4, 5, 6, 7, 8, 9,
                                         10, 11, 12, 13, 14, 15));
      v8us d0 = *(const v8us*)&sWl[coln][half * 8];
      v8us d1 = *(const v8us*)&sWl[coln][16 + half * 8];
      v16bf bl = __builtin_bit_cast(
          v16bf, __builtin_shufflevector(d0, d1, 0, 1, 2, 3, 4, 5, 6, 7, 8, 9,
                                         10, 11, 12, 13, 14, 15));
#pragma unroll
      for (int mt = 0; mt < 2; ++mt) {
        v8f a = acc[mt][nt];
        a = __builtin_amdgcn_wmma_f32_16x16x32_bf16(false, axh[mt], false, bh,
                                                    (short)0, a, false, false);
        a = __builtin_amdgcn_wmma_f32_16x16x32_bf16(false, axh[mt], false, bl,
                                                    (short)0, a, false, false);
        a = __builtin_amdgcn_wmma_f32_16x16x32_bf16(false, axl[mt], false, bh,
                                                    (short)0, a, false, false);
        acc[mt][nt] = a;
      }
    }
    __syncthreads();  // all waves done reading before next TDM overwrites LDS
  }

  // ---- store C (VGPR r: lanes 0-15 -> M=r, lanes 16-31 -> M=r+8)
#pragma unroll
  for (int mt = 0; mt < 2; ++mt) {
#pragma unroll
    for (int nt = 0; nt < 4; ++nt) {
      int gm = m0 + wave_m * 32 + mt * 16 + half * 8;
      int gn = n0 + wave_n * 64 + nt * 16 + lr;
#pragma unroll
      for (int r = 0; r < 8; ++r)
        Y[(size_t)(gm + r) * N_DIM + gn] = acc[mt][nt][r];
    }
  }
}

extern "C" void kernel_launch(void* const* d_in, const int* in_sizes, int n_in,
                              void* d_out, int out_size, void* d_ws,
                              size_t ws_size, hipStream_t stream) {
  (void)in_sizes; (void)n_in; (void)out_size; (void)ws_size;
  const float*         x     = (const float*)d_in[0];
  const float*         w     = (const float*)d_in[1];
  const float*         g     = (const float*)d_in[2];
  const unsigned char* mask  = (const unsigned char*)d_in[3];
  const int*           ndrop = (const int*)d_in[4];
  float*               out   = (float*)d_out;

  unsigned* u32ws = (unsigned*)d_ws;
  unsigned* hW1 = u32ws + OFF_HW1;
  unsigned* hW2 = u32ws + OFF_HW2;
  unsigned* hW3 = u32ws + OFF_HW3;
  unsigned* hG1 = u32ws + OFF_HG1;
  unsigned* hG2 = u32ws + OFF_HG2;
  unsigned* hG3 = u32ws + OFF_HG3;
  unsigned* res = u32ws + OFF_RES;
  unsigned short* whi = (unsigned short*)((char*)d_ws + OFF_PLANES);
  unsigned short* wlo = whi + TOTAL;

  hipMemsetAsync(d_ws, 0, 65536, stream);

  const int HB = 4096;  // histogram pass blocks
  // drop threshold: k-th smallest |w|  (3-level radix select, ascending)
  hist_level_kernel<<<HB, 256, 0, stream>>>(w, hW1, res, 2048, 1, 0, nullptr, nullptr);
  select_kernel<<<1, 32, 0, stream>>>(res, hW1, 2048, 0, ndrop, -1, 0, 1);
  hist_level_kernel<<<HB, 256, 0, stream>>>(w, hW2, res, 4096, 2, 0, nullptr, nullptr);
  select_kernel<<<1, 32, 0, stream>>>(res, hW2, 4096, 0, ndrop, 1, 2, 3);
  hist_level_kernel<<<HB, 256, 0, stream>>>(w, hW3, res, 256, 3, 0, nullptr, nullptr);
  select_kernel<<<1, 32, 0, stream>>>(res, hW3, 256, 0, ndrop, 3, 4, 5);
  // grow threshold: k-th largest |grad| among masked-out-after-drop slots
  hist_level_kernel<<<HB, 256, 0, stream>>>(g, hG1, res, 2048, 1, 6, w, mask);
  select_kernel<<<1, 32, 0, stream>>>(res, hG1, 2048, 1, ndrop, -1, 6, 7);
  hist_level_kernel<<<HB, 256, 0, stream>>>(g, hG2, res, 4096, 2, 6, w, mask);
  select_kernel<<<1, 32, 0, stream>>>(res, hG2, 4096, 1, ndrop, 7, 8, 9);
  hist_level_kernel<<<HB, 256, 0, stream>>>(g, hG3, res, 256, 3, 6, w, mask);
  select_kernel<<<1, 32, 0, stream>>>(res, hG3, 256, 1, ndrop, 9, 10, 11);
  // masked weight -> bf16 hi/lo planes
  finalize_kernel<<<8192, 256, 0, stream>>>(w, g, mask, res, whi, wlo);
  // compute-bound GEMM on the WMMA + TDM path
  dim3 grid(N_DIM / BN, M_DIM / BM);
  wmma_gemm_kernel<<<grid, 256, 0, stream>>>(x, whi, wlo, out);
}
